// MultiHeadAttention_9887014716153
// MI455X (gfx1250) — matrix operand order repaired and checked by reading, not compiled, not run on hardware
//
#include <hip/hip_runtime.h>

// ---------------------------------------------------------------------------
// MI455X (gfx1250) multi-head attention: bf16 WMMA everywhere, one-time f32->
// bf16 pre-conversion so GEMM inner loops are pure load + v_wmma.
// Softmax runs in the exp2 domain (v_exp_f32 is 2^x); log2(e) folded into the
// Q-projection scale. Attention processes 64 keys per block to amortize the
// per-row softmax VALU (shuffle reductions, m/l updates, O rescales).
//   D_MODEL=1024, N_HEADS=16, HEAD=64, B=4, T=1024
// ---------------------------------------------------------------------------

#define D_MODEL 1024
#define NHEADS  16
#define HEADD   64
#define BATCH   4
#define SEQ     1024
#define LOG2E   1.4426950408889634f

typedef __attribute__((ext_vector_type(16))) __bf16 bf16x16;
typedef __attribute__((ext_vector_type(8)))  __bf16 bf16x8;
typedef __attribute__((ext_vector_type(8)))  float  f32x8;

// D = A(16x32 bf16) * B(32x16 bf16) + C(16x16 f32)
#define WMMA_BF16(a, b, c) \
  __builtin_amdgcn_wmma_f32_16x16x32_bf16(false, (a), false, (b), (short)0, (c), false, false)

// ---- helpers --------------------------------------------------------------
__device__ __forceinline__ unsigned short f2bf(float f) {
  union { float f; unsigned u; } x; x.f = f;
  unsigned r = x.u + 0x7FFFu + ((x.u >> 16) & 1u);
  return (unsigned short)(r >> 16);
}
__device__ __forceinline__ float fast_exp2(float x) {
#if __has_builtin(__builtin_amdgcn_exp2f)
  return __builtin_amdgcn_exp2f(x);   // v_exp_f32
#else
  return exp2f(x);
#endif
}
__device__ __forceinline__ bf16x16 ldbf16(const unsigned short* p) {
  return *(const bf16x16*)p;   // 32B
}
__device__ __forceinline__ bf16x8 ldbf8(const unsigned short* p) {
  return *(const bf16x8*)p;    // 16B
}

// ---------------------------------------------------------------------------
// Kernel 0: one-time f32 -> bf16 conversion (row-major preserved).
// ---------------------------------------------------------------------------
__global__ __launch_bounds__(256) void cvt_f32_to_bf16(
    const float* __restrict__ src, unsigned short* __restrict__ dst, int n)
{
  const int idx = (blockIdx.x * 256 + threadIdx.x) * 8;
  if (idx >= n) return;
  const float4* p4 = (const float4*)(src + idx);
  const float4 a = p4[0], b = p4[1];
  union { unsigned short s[8]; uint4 v; } u;
  u.s[0] = f2bf(a.x); u.s[1] = f2bf(a.y); u.s[2] = f2bf(a.z); u.s[3] = f2bf(a.w);
  u.s[4] = f2bf(b.x); u.s[5] = f2bf(b.y); u.s[6] = f2bf(b.z); u.s[7] = f2bf(b.w);
  *(uint4*)(dst + idx) = u.v;
}

// ---------------------------------------------------------------------------
// Kernel 1: QKV projections.  out = X @ W^T, all operands bf16.
// One wave => 32(M) x 64(N): per K-step 2 A loads + 4 B loads feed 8 WMMA.
//   Qh/Kh layout: (b,h,t,d) bf16; Q pre-scaled by log2(e)/sqrt(HEAD).
//   V stored transposed: Vt (b,h,d,t) bf16 (t contiguous).
// ---------------------------------------------------------------------------
__global__ __launch_bounds__(256) void mha_proj_qkv(
    const unsigned short* __restrict__ Xq, const unsigned short* __restrict__ Xk,
    const unsigned short* __restrict__ Xv,
    const unsigned short* __restrict__ Wqb, const unsigned short* __restrict__ Wkb,
    const unsigned short* __restrict__ Wvb,
    unsigned short* __restrict__ Qh, unsigned short* __restrict__ Kh,
    unsigned short* __restrict__ Vt)
{
  const int w    = blockIdx.y;            // 0=Q 1=K 2=V
  const int wid  = threadIdx.x >> 5;
  const int lane = threadIdx.x & 31;
  const int li   = lane & 15;
  const int half = lane >> 4;
  const int task = blockIdx.x * 8 + wid;  // 2048 wave tasks
  const int nchunk = task & 15;
  const int mtile  = task >> 4;           // 128 tiles of 32 rows

  const unsigned short* X = (w == 0) ? Xq : (w == 1) ? Xk : Xv;
  const unsigned short* W = (w == 0) ? Wqb : (w == 1) ? Wkb : Wvb;

  const unsigned short* Xr0 = X + (size_t)(mtile * 32 + li) * D_MODEL;
  const unsigned short* Xr1 = Xr0 + 16 * D_MODEL;
  const int nbase = nchunk * 64;
  const unsigned short* Wr[4];
#pragma unroll
  for (int nt = 0; nt < 4; ++nt)
    Wr[nt] = W + (size_t)(nbase + nt * 16 + li) * D_MODEL;

  f32x8 acc[2][4] = {};
  for (int k0 = 0; k0 < D_MODEL; k0 += 32) {
    bf16x16 a0, a1;
    ((bf16x8*)&a0)[0] = ldbf8(Xr0 + k0 + half * 16);
    ((bf16x8*)&a0)[1] = ldbf8(Xr0 + k0 + half * 16 + 8);
    ((bf16x8*)&a1)[0] = ldbf8(Xr1 + k0 + half * 16);
    ((bf16x8*)&a1)[1] = ldbf8(Xr1 + k0 + half * 16 + 8);
    __builtin_prefetch((const void*)(Xr0 + k0 + 64), 0, 1);
    __builtin_prefetch((const void*)(Xr1 + k0 + 64), 0, 1);
    const int koff = k0 + half * 16;
#pragma unroll
    for (int nt = 0; nt < 4; ++nt) {
      const bf16x16 bt = ldbf16(Wr[nt] + koff);
      acc[0][nt] = WMMA_BF16(a0, bt, acc[0][nt]);
      acc[1][nt] = WMMA_BF16(a1, bt, acc[1][nt]);
    }
  }

  if (w < 2) {
    unsigned short* dst = (w == 0) ? Qh : Kh;
    const float scale = (w == 0) ? (0.125f * LOG2E) : 1.0f;  // exp2-domain scores
#pragma unroll
    for (int mt = 0; mt < 2; ++mt) {
#pragma unroll
      for (int nt = 0; nt < 4; ++nt) {
        const int n = nbase + nt * 16 + li;
        const int h = n >> 6, d = n & 63;
#pragma unroll
        for (int j = 0; j < 8; ++j) {
          const int m = mtile * 32 + mt * 16 + half * 8 + j;
          const int bb = m >> 10, t = m & (SEQ - 1);
          dst[((size_t)(bb * NHEADS + h) * SEQ + t) * HEADD + d] =
              f2bf(acc[mt][nt][j] * scale);
        }
      }
    }
  } else {
#pragma unroll
    for (int mt = 0; mt < 2; ++mt) {
      const int m0 = mtile * 32 + mt * 16 + half * 8;
      const int bb = m0 >> 10, t0 = m0 & (SEQ - 1);
#pragma unroll
      for (int nt = 0; nt < 4; ++nt) {
        const int n = nbase + nt * 16 + li;
        const int h = n >> 6, d = n & 63;
        union { unsigned short s[8]; uint4 v; } u;
#pragma unroll
        for (int j = 0; j < 8; ++j) u.s[j] = f2bf(acc[mt][nt][j]);
        *(uint4*)(Vt + ((size_t)(bb * NHEADS + h) * HEADD + d) * SEQ + t0) = u.v;
      }
    }
  }
}

// ---------------------------------------------------------------------------
// Kernel 2: flash attention, 64 keys per block.
// One wave = (b, h, 16 query rows), full HEAD=64. Online softmax in exp2
// domain; P tile transposed D-layout -> A-layout through LDS.
// ---------------------------------------------------------------------------
__global__ __launch_bounds__(256) void mha_attention(
    const unsigned short* __restrict__ Qh, const unsigned short* __restrict__ Kh,
    const unsigned short* __restrict__ Vt, const unsigned char* __restrict__ mask,
    unsigned short* __restrict__ Y)
{
  __shared__ unsigned short Plds[8][16 * 64];  // per-wave 16x64 bf16 prob tile

  const int wid  = threadIdx.x >> 5;
  const int lane = threadIdx.x & 31;
  const int li   = lane & 15;
  const int half = lane >> 4;
  const int task = blockIdx.x * 8 + wid;       // 4096 wave tasks
  const int qb = task & 63;
  const int h  = (task >> 6) & (NHEADS - 1);
  const int b  = task >> 10;

  // Q as two A tiles (d = 0..31, 32..63); pre-scaled by log2e/8.
  const unsigned short* Qbase =
      Qh + ((size_t)(b * NHEADS + h) * SEQ + qb * 16 + li) * HEADD;
  bf16x16 qa0, qa1;
  ((bf16x8*)&qa0)[0] = ldbf8(Qbase + half * 16);
  ((bf16x8*)&qa0)[1] = ldbf8(Qbase + half * 16 + 8);
  ((bf16x8*)&qa1)[0] = ldbf8(Qbase + 32 + half * 16);
  ((bf16x8*)&qa1)[1] = ldbf8(Qbase + 32 + half * 16 + 8);

  const unsigned short* Kbase = Kh + (size_t)(b * NHEADS + h) * SEQ * HEADD;
  const unsigned short* Vbase = Vt + (size_t)(b * NHEADS + h) * HEADD * SEQ;
  const unsigned char*  mk    = mask + (size_t)b * SEQ;

  f32x8 o[4] = {};
  float mrow[8], lrow[8];
#pragma unroll
  for (int j = 0; j < 8; ++j) { mrow[j] = -1e30f; lrow[j] = 0.0f; }

  unsigned short* Pw = &Plds[wid][0];

  for (int kb = 0; kb < SEQ; kb += 64) {
    // ---- scores S = Q K^T for 64 keys (four 16-col D tiles) ----
    f32x8 s[4];
#pragma unroll
    for (int c = 0; c < 4; ++c) {
      const unsigned short* Kc = Kbase + (size_t)(kb + c * 16 + li) * HEADD + half * 16;
      f32x8 t = {};
      t = WMMA_BF16(qa0, ldbf16(Kc), t);
      t = WMMA_BF16(qa1, ldbf16(Kc + 32), t);
      s[c] = t;
    }

    // prefetch next key block (speculative; addresses stay inside workspace)
    __builtin_prefetch((const void*)(Kbase + (size_t)(kb + 64 + li) * HEADD), 0, 1);
    __builtin_prefetch((const void*)(Vbase + (size_t)li * SEQ + kb + 64), 0, 1);

    float msk[4];
#pragma unroll
    for (int c = 0; c < 4; ++c)
      msk[c] = mk[kb + c * 16 + li] ? 0.0f : -1e30f;

    // ---- online softmax (row = 8*half + j lives on 16 lanes of one VGPR) ----
#pragma unroll
    for (int j = 0; j < 8; ++j) {
      const float v0 = s[0][j] + msk[0];
      const float v1 = s[1][j] + msk[1];
      const float v2 = s[2][j] + msk[2];
      const float v3 = s[3][j] + msk[3];
      float mx = fmaxf(fmaxf(v0, v1), fmaxf(v2, v3));
      mx = fmaxf(mx, __shfl_xor(mx, 1, 32));
      mx = fmaxf(mx, __shfl_xor(mx, 2, 32));
      mx = fmaxf(mx, __shfl_xor(mx, 4, 32));
      mx = fmaxf(mx, __shfl_xor(mx, 8, 32));
      const float nm    = fmaxf(mrow[j], mx);
      const float alpha = fast_exp2(mrow[j] - nm);
      const float p0 = fast_exp2(v0 - nm);
      const float p1 = fast_exp2(v1 - nm);
      const float p2 = fast_exp2(v2 - nm);
      const float p3 = fast_exp2(v3 - nm);
      float rs = (p0 + p1) + (p2 + p3);
      rs += __shfl_xor(rs, 1, 32);
      rs += __shfl_xor(rs, 2, 32);
      rs += __shfl_xor(rs, 4, 32);
      rs += __shfl_xor(rs, 8, 32);
      lrow[j] = lrow[j] * alpha + rs;
      mrow[j] = nm;
      o[0][j] *= alpha; o[1][j] *= alpha; o[2][j] *= alpha; o[3][j] *= alpha;
      // store P row-major (query row, key col); same-wave DS ops stay in order
      unsigned short* pr = Pw + (half * 8 + j) * 64 + li;
      pr[0]  = f2bf(p0);
      pr[16] = f2bf(p1);
      pr[32] = f2bf(p2);
      pr[48] = f2bf(p3);
    }

    // ---- re-read P in WMMA A layout (two 16x32 A tiles) ----
    const unsigned short* Pr = Pw + li * 64;
    bf16x16 pa0, pa1;
    ((bf16x8*)&pa0)[0] = *(const bf16x8*)(Pr + half * 16);
    ((bf16x8*)&pa0)[1] = *(const bf16x8*)(Pr + half * 16 + 8);
    ((bf16x8*)&pa1)[0] = *(const bf16x8*)(Pr + 32 + half * 16);
    ((bf16x8*)&pa1)[1] = *(const bf16x8*)(Pr + 32 + half * 16 + 8);

    // ---- O += P @ V (V transposed: 16 contiguous t per lane) ----
#pragma unroll
    for (int n = 0; n < 4; ++n) {
      const unsigned short* Vn = Vbase + (size_t)(n * 16 + li) * SEQ + kb + half * 16;
      o[n] = WMMA_BF16(pa0, ldbf16(Vn), o[n]);
      o[n] = WMMA_BF16(pa1, ldbf16(Vn + 32), o[n]);
    }
  }

  // ---- normalize, write Y (b, t, h*64+d) bf16 ----
#pragma unroll
  for (int j = 0; j < 8; ++j) {
    const float inv = 1.0f / lrow[j];
    const int t = qb * 16 + half * 8 + j;
    unsigned short* yp = Y + ((size_t)b * SEQ + t) * D_MODEL + h * HEADD + li;
    yp[0]  = f2bf(o[0][j] * inv);
    yp[16] = f2bf(o[1][j] * inv);
    yp[32] = f2bf(o[2][j] * inv);
    yp[48] = f2bf(o[3][j] * inv);
  }
}

// ---------------------------------------------------------------------------
// Kernel 3: output projection  out = Y @ Wo^T  (f32 result), 32x64 per wave.
// ---------------------------------------------------------------------------
__global__ __launch_bounds__(256) void mha_proj_out(
    const unsigned short* __restrict__ Y, const unsigned short* __restrict__ Wob,
    float* __restrict__ out)
{
  const int wid  = threadIdx.x >> 5;
  const int lane = threadIdx.x & 31;
  const int li   = lane & 15;
  const int half = lane >> 4;
  const int task = blockIdx.x * 8 + wid;  // 2048
  const int nchunk = task & 15;
  const int mtile  = task >> 4;

  const unsigned short* Yr0 = Y + (size_t)(mtile * 32 + li) * D_MODEL;
  const unsigned short* Yr1 = Yr0 + 16 * D_MODEL;
  const int nbase = nchunk * 64;
  const unsigned short* Wr[4];
#pragma unroll
  for (int nt = 0; nt < 4; ++nt)
    Wr[nt] = Wob + (size_t)(nbase + nt * 16 + li) * D_MODEL;

  f32x8 acc[2][4] = {};
  for (int k0 = 0; k0 < D_MODEL; k0 += 32) {
    bf16x16 a0, a1;
    ((bf16x8*)&a0)[0] = ldbf8(Yr0 + k0 + half * 16);
    ((bf16x8*)&a0)[1] = ldbf8(Yr0 + k0 + half * 16 + 8);
    ((bf16x8*)&a1)[0] = ldbf8(Yr1 + k0 + half * 16);
    ((bf16x8*)&a1)[1] = ldbf8(Yr1 + k0 + half * 16 + 8);
    __builtin_prefetch((const void*)(Yr0 + k0 + 64), 0, 1);
    __builtin_prefetch((const void*)(Yr1 + k0 + 64), 0, 1);
    const int koff = k0 + half * 16;
#pragma unroll
    for (int nt = 0; nt < 4; ++nt) {
      const bf16x16 bt = ldbf16(Wr[nt] + koff);
      acc[0][nt] = WMMA_BF16(a0, bt, acc[0][nt]);
      acc[1][nt] = WMMA_BF16(a1, bt, acc[1][nt]);
    }
  }

#pragma unroll
  for (int mt = 0; mt < 2; ++mt) {
#pragma unroll
    for (int nt = 0; nt < 4; ++nt) {
      const int n = nbase + nt * 16 + li;
#pragma unroll
      for (int j = 0; j < 8; ++j) {
        const int m = mtile * 32 + mt * 16 + half * 8 + j;
        out[(size_t)m * D_MODEL + n] = acc[mt][nt][j];
      }
    }
  }
}

// ---------------------------------------------------------------------------
extern "C" void kernel_launch(void* const* d_in, const int* in_sizes, int n_in,
                              void* d_out, int out_size, void* d_ws, size_t ws_size,
                              hipStream_t stream) {
  const float* q  = (const float*)d_in[0];
  const float* k  = (const float*)d_in[1];
  const float* v  = (const float*)d_in[2];
  const unsigned char* mask = (const unsigned char*)d_in[3];
  const float* Wq = (const float*)d_in[4];
  const float* Wk = (const float*)d_in[5];
  const float* Wv = (const float*)d_in[6];
  const float* Wo = (const float*)d_in[7];

  // workspace (bf16 elems): Qh|Kh|Vt|Yb (4E) + Xq|Xk|Xv (3E) + 4 weights (4W)
  // = 32M bf16 = 64 MB
  const size_t E  = (size_t)BATCH * SEQ * D_MODEL;
  const size_t Wn = (size_t)D_MODEL * D_MODEL;
  unsigned short* Qh  = (unsigned short*)d_ws;
  unsigned short* Kh  = Qh + E;
  unsigned short* Vt  = Kh + E;
  unsigned short* Yb  = Vt + E;
  unsigned short* Xq  = Yb + E;
  unsigned short* Xk  = Xq + E;
  unsigned short* Xv  = Xk + E;
  unsigned short* Wqb = Xv + E;
  unsigned short* Wkb = Wqb + Wn;
  unsigned short* Wvb = Wkb + Wn;
  unsigned short* Wob = Wvb + Wn;

  const int xblk = (int)(E / 2048);
  const int wblk = (int)(Wn / 2048);
  cvt_f32_to_bf16<<<xblk, 256, 0, stream>>>(q,  Xq,  (int)E);
  cvt_f32_to_bf16<<<xblk, 256, 0, stream>>>(k,  Xk,  (int)E);
  cvt_f32_to_bf16<<<xblk, 256, 0, stream>>>(v,  Xv,  (int)E);
  cvt_f32_to_bf16<<<wblk, 256, 0, stream>>>(Wq, Wqb, (int)Wn);
  cvt_f32_to_bf16<<<wblk, 256, 0, stream>>>(Wk, Wkb, (int)Wn);
  cvt_f32_to_bf16<<<wblk, 256, 0, stream>>>(Wv, Wvb, (int)Wn);
  cvt_f32_to_bf16<<<wblk, 256, 0, stream>>>(Wo, Wob, (int)Wn);

  mha_proj_qkv<<<dim3(256, 3), 256, 0, stream>>>(Xq, Xk, Xv, Wqb, Wkb, Wvb, Qh, Kh, Vt);
  mha_attention<<<512, 256, 0, stream>>>(Qh, Kh, Vt, mask, Yb);
  mha_proj_out<<<256, 256, 0, stream>>>(Yb, Wob, (float*)d_out);
}
